// LSTM_2052994367852
// MI455X (gfx1250) — compile-verified, matrix operand order checked by reading, not run on hardware
//
#include <hip/hip_runtime.h>
#include <hip/hip_bf16.h>

typedef __attribute__((ext_vector_type(16))) _Float16 v16h;
typedef __attribute__((ext_vector_type(8)))  float    v8f;

#define BATCH 512
#define HIST  8
#define PRED  24
#define NCITY 184
#define FEAT  13
#define HID   32
#define BCSEQ (BATCH * NCITY)   // 94208 = 16 * 5888

__device__ __forceinline__ float fsig(float x) {
    // 1 / (1 + 2^(-x*log2e))
    return __builtin_amdgcn_rcpf(1.0f + __builtin_amdgcn_exp2f(-1.4426950408889634f * x));
}
__device__ __forceinline__ float ftanh(float x) {
    // 1 - 2 / (2^(2x*log2e) + 1)
    return 1.0f - 2.0f * __builtin_amdgcn_rcpf(1.0f + __builtin_amdgcn_exp2f(2.8853900817779268f * x));
}

// ---------------------------------------------------------------------------
// Prep kernel (1 block, 256 threads): fold the input projection into the
// recurrent path.
//   U     = W_ih @ W_in                  (128 x 14)
//   Wr    = W_hh + outer(U[:,0], W_out)  (128 x 32)   combined recurrent matrix
//   biasC = b_ih + b_hh + W_ih@b_in + b_out*U[:,0]
// Then emit f16 WMMA B-fragments (32x16 per tile, 8 tiles of 16 gate columns)
// for Wr^T and U[:,1:14]^T (K padded to 32 with zeros).
// B-frag layout (wave32, 16-bit): lane l<16 -> col N=l, halves {K0..7,K16..23};
// lane l>=16 -> col N=l-16, halves {K8..15,K24..31}.
// ---------------------------------------------------------------------------
__global__ void lstm_prep(const float* __restrict__ W_in,  const float* __restrict__ b_in,
                          const float* __restrict__ W_out, const float* __restrict__ b_out,
                          const float* __restrict__ W_ih,  const float* __restrict__ W_hh,
                          const float* __restrict__ b_ih,  const float* __restrict__ b_hh,
                          float* __restrict__ Wr, float* __restrict__ U,
                          float* __restrict__ u0, float* __restrict__ biasC,
                          _Float16* __restrict__ WrB, _Float16* __restrict__ UB)
{
    const int tid = threadIdx.x;
    if (tid < 128) {
        float Urow[FEAT + 1];
        #pragma unroll
        for (int c = 0; c < FEAT + 1; ++c) {
            float s = 0.f;
            for (int k = 0; k < HID; ++k) s += W_ih[tid * HID + k] * W_in[k * (FEAT + 1) + c];
            Urow[c] = s;
            U[tid * (FEAT + 1) + c] = s;
        }
        const float u0v = Urow[0];
        u0[tid] = u0v;
        float bc = b_ih[tid] + b_hh[tid] + b_out[0] * u0v;
        for (int k = 0; k < HID; ++k) bc += W_ih[tid * HID + k] * b_in[k];
        biasC[tid] = bc;
        for (int k = 0; k < HID; ++k)
            Wr[tid * HID + k] = W_hh[tid * HID + k] + u0v * W_out[k];
    }
    __syncthreads();
    // tid = n*32 + l : one (tile n, lane-slot l) B-fragment of 16 halves
    {
        const int n = tid >> 5, l = tid & 31;
        const int col = 16 * n + (l & 15);
        const int khalf = (l >= 16) ? 8 : 0;
        #pragma unroll
        for (int k = 0; k < 16; ++k) {
            const int K = (k & 7) + khalf + ((k >= 8) ? 16 : 0);
            WrB[tid * 16 + k] = (_Float16)Wr[col * HID + K];
            const float uv = (K < FEAT) ? U[col * (FEAT + 1) + (K + 1)] : 0.f;
            UB[tid * 16 + k] = (_Float16)uv;
        }
    }
}

// ---------------------------------------------------------------------------
// Main kernel: one wave32 = one tile of 16 sequences, full 24-step rollout.
// Per step per wave: 16 x v_wmma_f32_16x16x32_f16 (8 feat-term, 8 recurrent),
// VALU LSTM elementwise in C-layout, shfl-reduce output head, LDS re-swizzle
// of h into A-fragment layout for the next step.
// ---------------------------------------------------------------------------
__global__ void __launch_bounds__(256) lstm_main(
    const float* __restrict__ pm25, const float* __restrict__ feature,
    const float* __restrict__ W_out, const float* __restrict__ b_out_p,
    const float* __restrict__ u0, const float* __restrict__ biasC,
    const _Float16* __restrict__ WrB, const _Float16* __restrict__ UB,
    float* __restrict__ out)
{
    const int lane  = threadIdx.x & 31;
    const int wv    = threadIdx.x >> 5;
    const int tile  = blockIdx.x * 8 + wv;
    const int seq0  = tile * 16;
    const int r     = lane & 15;          // A-row / B,C-col index within tile
    const bool hi   = lane >= 16;
    const int mbase = hi ? 8 : 0;         // C-layout row base for this half-wave

    __shared__ _Float16 hbuf[8][16][HID]; // per-wave h re-swizzle buffer
    __shared__ float    x0buf[8][16];

    // resident weight fragments (f16 B-layout)
    v16h wrb[8], ub[8];
    #pragma unroll
    for (int n = 0; n < 8; ++n) {
        wrb[n] = *(const v16h*)(WrB + (n * 32 + lane) * 16);
        ub[n]  = *(const v16h*)(UB  + (n * 32 + lane) * 16);
    }
    float biasn[8], u0n[8];
    #pragma unroll
    for (int n = 0; n < 8; ++n) {
        biasn[n] = biasC[16 * n + r];
        u0n[n]   = u0[16 * n + r];
    }
    const float wa = W_out[r];
    const float wb = W_out[16 + r];
    const float bo = b_out_p[0];

    // broadcast x0 (= pm25_hist[:, -1]) for the 16 rows of this tile
    if (!hi) {
        const int row = seq0 + lane;
        const int b = row / NCITY, city = row - b * NCITY;
        x0buf[wv][lane] = pm25[(b * HIST + (HIST - 1)) * NCITY + city];
    }
    __syncthreads();
    float x0m[8];
    #pragma unroll
    for (int m = 0; m < 8; ++m) x0m[m] = x0buf[wv][mbase + m] - bo;

    // per-lane feature base pointer (A-fragment row = seq0 + r)
    const int arow = seq0 + r;
    const int ab = arow / NCITY, acity = arow - ab * NCITY;
    const float* fbase = feature + ((size_t)(ab * (HIST + PRED) + HIST) * NCITY + acity) * FEAT;

    v8f c0v = (v8f)(0.0f), c1v = (v8f)(0.0f);
    v16h hA;
    #pragma unroll
    for (int i = 0; i < 16; ++i) hA[i] = (_Float16)0.0f;

    for (int t = 0; t < PRED; ++t) {
        const float* fp = fbase + (size_t)t * (NCITY * FEAT);

        // feat -> A-fragment (K = 0..12 valid, padded to 32)
        v16h aF;
        if (!hi) {
            #pragma unroll
            for (int k = 0; k < 8; ++k)  aF[k] = (_Float16)fp[k];       // K 0..7
            #pragma unroll
            for (int k = 8; k < 16; ++k) aF[k] = (_Float16)0.0f;        // K 16..23
        } else {
            #pragma unroll
            for (int k = 0; k < 5; ++k)  aF[k] = (_Float16)fp[8 + k];   // K 8..12
            #pragma unroll
            for (int k = 5; k < 16; ++k) aF[k] = (_Float16)0.0f;        // pad
        }
        if (t + 1 < PRED) __builtin_prefetch(fp + NCITY * FEAT, 0, 3);

        // gates = bias (+ rank-1 x0 fix at t=0) + feat@U14^T + h@Wr^T
        v8f acc[8];
        #pragma unroll
        for (int n = 0; n < 8; ++n) {
            v8f a;
            if (t == 0) {
                #pragma unroll
                for (int m = 0; m < 8; ++m) a[m] = biasn[n] + x0m[m] * u0n[n];
            } else {
                #pragma unroll
                for (int m = 0; m < 8; ++m) a[m] = biasn[n];
            }
            a = __builtin_amdgcn_wmma_f32_16x16x32_f16(false, aF, false, ub[n],
                                                       (short)0, a, false, false);
            a = __builtin_amdgcn_wmma_f32_16x16x32_f16(false, hA, false, wrb[n],
                                                       (short)0, a, false, false);
            acc[n] = a;
        }

        // LSTM elementwise (C-layout; tiles 0..7 = i,i,f,f,g,g,o,o)
        v8f h0v, h1v;
        #pragma unroll
        for (int m = 0; m < 8; ++m) {
            const float ig0 = fsig(acc[0][m]),  ig1 = fsig(acc[1][m]);
            const float fg0 = fsig(acc[2][m]),  fg1 = fsig(acc[3][m]);
            const float gg0 = ftanh(acc[4][m]), gg1 = ftanh(acc[5][m]);
            const float og0 = fsig(acc[6][m]),  og1 = fsig(acc[7][m]);
            const float c0n = fg0 * c0v[m] + ig0 * gg0;
            const float c1n = fg1 * c1v[m] + ig1 * gg1;
            c0v[m] = c0n; c1v[m] = c1n;
            h0v[m] = og0 * ftanh(c0n);
            h1v[m] = og1 * ftanh(c1n);
        }

        // output head: ys[row] = h . w_out + b_out (cross-lane reduce over 32 cols)
        float part[8];
        #pragma unroll
        for (int m = 0; m < 8; ++m) part[m] = h0v[m] * wa + h1v[m] * wb;
        #pragma unroll
        for (int mask = 1; mask < 16; mask <<= 1) {
            #pragma unroll
            for (int m = 0; m < 8; ++m) part[m] += __shfl_xor(part[m], mask, 32);
        }
        if (r == 0) {
            #pragma unroll
            for (int m = 0; m < 8; ++m) {
                const int row = seq0 + mbase + m;
                const int b = row / NCITY, city = row - b * NCITY;
                out[(b * PRED + t) * NCITY + city] = part[m] + bo;
            }
        }

        // re-swizzle h (f32 C-layout) -> f16 A-fragment via per-wave LDS
        __syncthreads();
        #pragma unroll
        for (int m = 0; m < 8; ++m) {
            hbuf[wv][mbase + m][r]      = (_Float16)h0v[m];
            hbuf[wv][mbase + m][16 + r] = (_Float16)h1v[m];
        }
        __syncthreads();
        {
            const _Float16* hr = &hbuf[wv][r][0];
            const int koff = hi ? 8 : 0;
            #pragma unroll
            for (int k = 0; k < 8; ++k) hA[k]     = hr[koff + k];        // K {0..7|8..15}
            #pragma unroll
            for (int k = 0; k < 8; ++k) hA[8 + k] = hr[16 + koff + k];   // K {16..23|24..31}
        }
    }
}

extern "C" void kernel_launch(void* const* d_in, const int* in_sizes, int n_in,
                              void* d_out, int out_size, void* d_ws, size_t ws_size,
                              hipStream_t stream) {
    (void)in_sizes; (void)n_in; (void)out_size; (void)ws_size;
    const float* pm25    = (const float*)d_in[0];
    const float* feature = (const float*)d_in[1];
    const float* W_in    = (const float*)d_in[2];
    const float* b_in    = (const float*)d_in[3];
    const float* W_out   = (const float*)d_in[4];
    const float* b_out   = (const float*)d_in[5];
    const float* W_ih    = (const float*)d_in[6];
    const float* W_hh    = (const float*)d_in[7];
    const float* b_ih    = (const float*)d_in[8];
    const float* b_hh    = (const float*)d_in[9];
    float* out = (float*)d_out;

    char* ws = (char*)d_ws;
    float*    Wr    = (float*)ws;     ws += 128 * 32 * sizeof(float);   // 16384
    float*    U     = (float*)ws;     ws += 128 * 14 * sizeof(float);   // 7168
    float*    u0    = (float*)ws;     ws += 128 * sizeof(float);        // 512
    float*    biasC = (float*)ws;     ws += 128 * sizeof(float);        // 512
    _Float16* WrB   = (_Float16*)ws;  ws += 8 * 32 * 16 * sizeof(_Float16); // 8192
    _Float16* UB    = (_Float16*)ws;  ws += 8 * 32 * 16 * sizeof(_Float16); // 8192

    lstm_prep<<<dim3(1), dim3(256), 0, stream>>>(W_in, b_in, W_out, b_out,
                                                 W_ih, W_hh, b_ih, b_hh,
                                                 Wr, U, u0, biasC, WrB, UB);

    // BCSEQ/16 = 5888 tiles, 8 waves (tiles) per 256-thread block -> 736 blocks
    lstm_main<<<dim3(BCSEQ / 16 / 8), dim3(256), 0, stream>>>(
        pm25, feature, W_out, b_out, u0, biasC, WrB, UB, out);
}
